// GlobalDecoderLayer_48876727829095
// MI455X (gfx1250) — matrix operand order repaired and checked by reading, not compiled, not run on hardware
//
#include <hip/hip_runtime.h>
#include <hip/hip_fp16.h>

typedef __attribute__((ext_vector_type(16))) _Float16 v16h;
typedef __attribute__((ext_vector_type(8)))  float    v8f;

#define NQ_  512
#define B_   2
#define NK_  2048
#define C_   256
#define NH_  8
#define HD_  32
#define DFF_ 2048
#define LN_EPS 1e-5f
#define LPAD 36   // LDS row stride (floats) for 32-float tile rows: kills bank conflicts

// ===================== small-tile WMMA GEMM (thin N / transB=0 cases) =====================
__global__ void gemm_wmma(
    const float* __restrict__ A, int lda, long long sAb, long long sAh,
    const float* __restrict__ Bm, int ldb, int transB, long long sBb, long long sBh,
    const float* __restrict__ bias,
    float* __restrict__ out, int ldo, long long sOb, long long sOh,
    const float* __restrict__ resid, int res_mode, int nq,
    int M, int N, int K, float alpha, int do_relu, int nh)
{
  const int wave = threadIdx.x >> 5;
  const int lane = threadIdx.x & 31;
  const int tiles_n = N >> 4;
  const int tiles   = (M >> 4) * tiles_n;
  const int tile = blockIdx.x * (blockDim.x >> 5) + wave;
  if (tile >= tiles) return;                 // wave-uniform, EXEC stays all-ones
  const int m0 = (tile / tiles_n) << 4;
  const int n0 = (tile % tiles_n) << 4;
  const int bi = blockIdx.y;
  const int bb = bi / nh, hh = bi % nh;
  const float* Ab = A  + bb * sAb + hh * sAh;
  const float* Bb = Bm + bb * sBb + hh * sBh;
  const int r  = lane & 15;
  const int hi = lane >> 4;

  v8f acc = {};
  for (int k0 = 0; k0 < K; k0 += 32) {
    v16h af, bf;
    const float* ap = Ab + (size_t)(m0 + r) * lda + (k0 + 16 * hi);
#pragma unroll
    for (int i = 0; i < 8; ++i) { af[i] = (_Float16)ap[i]; af[8 + i] = (_Float16)ap[8 + i]; }
    if (transB) {
      const float* bp = Bb + (size_t)(n0 + r) * ldb + (k0 + 16 * hi);
#pragma unroll
      for (int i = 0; i < 16; ++i) bf[i] = (_Float16)bp[i];
    } else {
      const float* bp = Bb + (size_t)(k0 + 16 * hi) * ldb + (n0 + r);
#pragma unroll
      for (int i = 0; i < 16; ++i) bf[i] = (_Float16)bp[(size_t)i * ldb];
    }
    if (k0 + 32 < K) __builtin_prefetch(ap + 32, 0, 3);   // global_prefetch_b8
    acc = __builtin_amdgcn_wmma_f32_16x16x32_f16(false, af, false, bf,
                                                 (short)0, acc, false, false);
  }

  const int n_g = n0 + r;
  const float bv = bias ? bias[n_g] : 0.0f;
#pragma unroll
  for (int j = 0; j < 8; ++j) {
    const int m_g = m0 + 8 * hi + j;
    float v = acc[j] + bv;
    if (do_relu) v = v > 0.0f ? v : 0.0f;
    v *= alpha;
    if (res_mode == 0) {
      out[bb * sOb + hh * sOh + (size_t)m_g * ldo + n_g] = v;
    } else {
      const int b2 = m_g / nq, q2 = m_g % nq;
      const size_t oi = ((size_t)q2 * (M / nq) + b2) * ldo + n_g;
      out[oi] = resid[oi] + v;
    }
  }
}

// ===================== 128x64-tile WMMA GEMM, double-buffered async-to-LDS =====================
// transB=1 only: out = epilogue(A[M,K] x W[N,K]^T + bias). M%128==0, N%64==0, K%32==0.
// 256 threads = 8 waves; wave owns a 2x2 grid of 16x16 subtiles -> 4 WMMAs/K-step with
// A/B fragment reuse. K-slabs staged with GLOBAL_LOAD_ASYNC_TO_LDS_B128 into 2 LDS
// buffers; next slab issued before computing current (ASYNCcnt in-order completion).
__global__ void gemm_wmma_tiled(
    const float* __restrict__ A, int lda, long long sAb, long long sAh,
    const float* __restrict__ Bm, int ldb, long long sBb, long long sBh,
    const float* __restrict__ bias,
    float* __restrict__ out, int ldo, long long sOb, long long sOh,
    const float* __restrict__ resid, int res_mode, int nq,
    int M, int N, int K, float alpha, int do_relu, int nh)
{
  __shared__ float sA[2][128 * LPAD];
  __shared__ float sB[2][64 * LPAD];
  const int wave = threadIdx.x >> 5;
  const int lane = threadIdx.x & 31;
  const int r  = lane & 15;
  const int hi = lane >> 4;
  const int tiles_n = N >> 6;
  const int m0 = (blockIdx.x / tiles_n) << 7;
  const int n0 = (blockIdx.x % tiles_n) << 6;
  const int bi = blockIdx.y;
  const int bb = bi / nh, hh = bi % nh;
  const float* Ab = A  + bb * sAb + hh * sAh;
  const float* Bb = Bm + bb * sBb + hh * sBh;

  const int sm0 = (wave & 3) << 1;       // first of two m-subtiles (0..7)
  const int sn0 = (wave >> 2) << 1;      // first of two n-subtiles (0..3)

  // stage one 128x32 A slab + 64x32 B slab (f32) into LDS buffer `buf`
  auto stage = [&](int k0, int buf) {
    const unsigned aBase = (unsigned)(uintptr_t)(void*)&sA[buf][0];
    const unsigned bBase = (unsigned)(uintptr_t)(void*)&sB[buf][0];
#pragma unroll
    for (int it = 0; it < 4; ++it) {                 // A: 1024 quads
      const int idx = threadIdx.x + (it << 8);
      const int row = idx >> 3, seg = idx & 7;
      const unsigned la = aBase + (unsigned)((row * LPAD + seg * 4) * 4);
      const unsigned long long ga =
          (unsigned long long)(uintptr_t)(Ab + (size_t)(m0 + row) * lda + (k0 + seg * 4));
      asm volatile("global_load_async_to_lds_b128 %0, %1, off"
                   :: "v"(la), "v"(ga) : "memory");
    }
#pragma unroll
    for (int it = 0; it < 2; ++it) {                 // B: 512 quads
      const int idx = threadIdx.x + (it << 8);
      const int row = idx >> 3, seg = idx & 7;
      const unsigned lb = bBase + (unsigned)((row * LPAD + seg * 4) * 4);
      const unsigned long long gb =
          (unsigned long long)(uintptr_t)(Bb + (size_t)(n0 + row) * ldb + (k0 + seg * 4));
      asm volatile("global_load_async_to_lds_b128 %0, %1, off"
                   :: "v"(lb), "v"(gb) : "memory");
    }
  };

  v8f acc[2][2];
#pragma unroll
  for (int u = 0; u < 2; ++u)
#pragma unroll
    for (int v = 0; v < 2; ++v) acc[u][v] = (v8f){};

  const int nk = K >> 5;
  stage(0, 0);
  for (int ks = 0; ks < nk; ++ks) {
    const int cur = ks & 1;
    if (ks + 1 < nk) {
      stage((ks + 1) << 5, cur ^ 1);                 // prefetch next slab (other buffer)
      asm volatile("s_wait_asynccnt 6" ::: "memory"); // current slab's 6 ops retired
    } else {
      asm volatile("s_wait_asynccnt 0" ::: "memory");
    }
    __syncthreads();                                  // slab visible to all waves

    v16h af[2], bf[2];
#pragma unroll
    for (int u = 0; u < 2; ++u) {
      const float* ap = &sA[cur][((sm0 + u) * 16 + r) * LPAD + 16 * hi];
#pragma unroll
      for (int i = 0; i < 8; ++i) { af[u][i] = (_Float16)ap[i]; af[u][8 + i] = (_Float16)ap[8 + i]; }
      const float* bp = &sB[cur][((sn0 + u) * 16 + r) * LPAD + 16 * hi];
#pragma unroll
      for (int i = 0; i < 16; ++i) bf[u][i] = (_Float16)bp[i];
    }
#pragma unroll
    for (int u = 0; u < 2; ++u)
#pragma unroll
      for (int v = 0; v < 2; ++v)
        acc[u][v] = __builtin_amdgcn_wmma_f32_16x16x32_f16(false, af[u], false, bf[v],
                                                           (short)0, acc[u][v], false, false);
    __syncthreads();                                  // reads done before buffer re-staged
  }

  // ---- epilogue: bias / relu / alpha / (transposed) residual store ----
#pragma unroll
  for (int u = 0; u < 2; ++u)
#pragma unroll
  for (int v = 0; v < 2; ++v) {
    const int n_g = n0 + (sn0 + v) * 16 + r;
    const float bv = bias ? bias[n_g] : 0.0f;
#pragma unroll
    for (int j = 0; j < 8; ++j) {
      const int m_g = m0 + (sm0 + u) * 16 + 8 * hi + j;
      float vv = acc[u][v][j] + bv;
      if (do_relu) vv = vv > 0.0f ? vv : 0.0f;
      vv *= alpha;
      if (res_mode == 0) {
        out[bb * sOb + hh * sOh + (size_t)m_g * ldo + n_g] = vv;
      } else {
        const int b2 = m_g / nq, q2 = m_g % nq;
        const size_t oi = ((size_t)q2 * (M / nq) + b2) * ldo + n_g;
        out[oi] = resid[oi] + vv;
      }
    }
  }
}

// ===================== LayerNorm (+query_pos, with NQB->BNQ transpose) =====================
__global__ void layernorm_qk(const float* __restrict__ x,
                             const float* __restrict__ g, const float* __restrict__ be,
                             const float* __restrict__ qpos,
                             float* __restrict__ t2, float* __restrict__ qk)
{
  __shared__ float sm[256];
  const int row = blockIdx.x;            // b*NQ + q
  const int b = row / NQ_, q = row % NQ_;
  const size_t src = ((size_t)q * B_ + b) * C_;
  const int c = threadIdx.x;
  const float v = x[src + c];
  sm[c] = v; __syncthreads();
  for (int s = 128; s > 0; s >>= 1) { if (c < s) sm[c] += sm[c + s]; __syncthreads(); }
  const float mean = sm[0] * (1.0f / C_); __syncthreads();
  const float dv = v - mean;
  sm[c] = dv * dv; __syncthreads();
  for (int s = 128; s > 0; s >>= 1) { if (c < s) sm[c] += sm[c + s]; __syncthreads(); }
  const float var = sm[0] * (1.0f / C_);
  const float nrm = dv * rsqrtf(var + LN_EPS) * g[c] + be[c];
  t2[(size_t)row * C_ + c] = nrm;
  if (qk) qk[(size_t)row * C_ + c] = nrm + qpos[src + c];
}

// ===================== row softmax (in place) =====================
__global__ void softmax_rows(float* __restrict__ d, int rowlen)
{
  __shared__ float sm[256];
  float* p = d + (size_t)blockIdx.x * rowlen;
  const int t = threadIdx.x;
  float mx = -3.4e38f;
  for (int i = t; i < rowlen; i += 256) mx = fmaxf(mx, p[i]);
  sm[t] = mx; __syncthreads();
  for (int s = 128; s > 0; s >>= 1) { if (t < s) sm[t] = fmaxf(sm[t], sm[t + s]); __syncthreads(); }
  mx = sm[0]; __syncthreads();
  float sum = 0.0f;
  for (int i = t; i < rowlen; i += 256) { const float e = expf(p[i] - mx); p[i] = e; sum += e; }
  sm[t] = sum; __syncthreads();
  for (int s = 128; s > 0; s >>= 1) { if (t < s) sm[t] += sm[t + s]; __syncthreads(); }
  const float inv = 1.0f / sm[0];
  __syncthreads();
  for (int i = t; i < rowlen; i += 256) p[i] *= inv;
}

// ===================== memory (NK,B,C) -> (B*NK,C) transpose =====================
__global__ void transpose_mem(const float* __restrict__ in, float* __restrict__ out)
{
  const size_t gid = (size_t)blockIdx.x * 256 + threadIdx.x;
  const int c = gid & (C_ - 1);
  const size_t t = gid / C_;
  const int kk = t % NK_;
  const int b  = t / NK_;
  out[gid] = in[((size_t)kk * B_ + b) * C_ + c];
}

// ===================== per-(b,q) cos/sin table =====================
__global__ void cos_sin_kernel(const float* __restrict__ ang, float* __restrict__ cs)
{
  const int idx = blockIdx.x * 256 + threadIdx.x;
  if (idx >= B_ * NQ_) return;
  cs[2 * idx + 0] = cosf(ang[idx]);
  cs[2 * idx + 1] = sinf(ang[idx]);
}

// ===================== CPB-MLP: build rpe_tables (8,NH,32,32,32) =====================
__global__ void cpb_kernel(const float* __restrict__ w1, const float* __restrict__ b1,
                           const float* __restrict__ w2, float* __restrict__ rt)
{
  __shared__ float sw1[384], sb1[128], sw2[1024];
  const int i = blockIdx.y;
  for (int t = threadIdx.x; t < 384;  t += 256) sw1[t] = w1[i * 384 + t];
  for (int t = threadIdx.x; t < 128;  t += 256) sb1[t] = b1[i * 128 + t];
  for (int t = threadIdx.x; t < 1024; t += 256) sw2[t] = w2[i * 1024 + t];
  __syncthreads();
  const int v = blockIdx.x * 256 + threadIdx.x;
  const int p  = v >> 10, qy = (v >> 5) & 31, rx = v & 31;
  const float X = -4.0f + 8.0f * p  / 31.0f;
  const float Y = -4.0f + 8.0f * qy / 31.0f;
  const float Z = -4.0f + 8.0f * rx / 31.0f;
  float acc[8];
#pragma unroll
  for (int ch = 0; ch < 8; ++ch) acc[ch] = 0.0f;
  for (int d = 0; d < 128; ++d) {
    float h = sw1[d * 3 + 0] * X + sw1[d * 3 + 1] * Y + sw1[d * 3 + 2] * Z + sb1[d];
    h = h > 0.0f ? h : 0.0f;
#pragma unroll
    for (int ch = 0; ch < 8; ++ch) acc[ch] += h * sw2[ch * 128 + d];
  }
#pragma unroll
  for (int ch = 0; ch < 8; ++ch) rt[((size_t)i * 8 + ch) * 32768 + v] = acc[ch];
}

// ===================== RPE: trilinear grid-sample bias, added to CA scores =====================
__global__ void rpe_add(float* __restrict__ scores,
                        const float* __restrict__ refp,
                        const float* __restrict__ cs,
                        const float* __restrict__ xyz,
                        const float* __restrict__ rt)
{
  const size_t gid = (size_t)blockIdx.x * 256 + threadIdx.x;  // exactly B*NQ*NK
  const int kk = (int)(gid % NK_);
  const int q  = (int)((gid / NK_) % NQ_);
  const int b  = (int)(gid / ((size_t)NK_ * NQ_));
  const size_t eo = ((size_t)b * NK_ + kk) * 3;
  const float ex = xyz[eo + 0], ey = xyz[eo + 1], ez = xyz[eo + 2];
  const size_t co = 2 * ((size_t)b * NQ_ + q);
  const float cth = cs[co + 0], sth = cs[co + 1];
  float acc[8];
#pragma unroll
  for (int ch = 0; ch < 8; ++ch) acc[ch] = 0.0f;

  for (int i = 0; i < 8; ++i) {
    const size_t rpo = (((size_t)b * NQ_ + q) * 8 + i) * 3;
    const float d0x = refp[rpo + 0] - ex;
    const float d0y = refp[rpo + 1] - ey;
    const float d0z = refp[rpo + 2] - ez;
    const float ax = d0x, ay = -d0z, az = d0y;
    const float rx = ax * cth - az * sth;
    const float ry = ay;
    const float rz = ax * sth + az * cth;
    const float dx = copysignf(log2f(fabsf(rx) * 512.0f + 1.0f) * (1.0f / 3.0f), rx);
    const float dy = copysignf(log2f(fabsf(rz) * 512.0f + 1.0f) * (1.0f / 3.0f), rz);
    const float mz = -ry;
    const float dz = copysignf(log2f(fabsf(mz) * 512.0f + 1.0f) * (1.0f / 3.0f), mz);
    const float xf = ((dx * 0.25f + 1.0f) * 32.0f - 1.0f) * 0.5f;
    const float yf = ((dy * 0.25f + 1.0f) * 32.0f - 1.0f) * 0.5f;
    const float zf = ((dz * 0.25f + 1.0f) * 32.0f - 1.0f) * 0.5f;
    const float x0f = floorf(xf), y0f = floorf(yf), z0f = floorf(zf);
    const float fx = xf - x0f, fy = yf - y0f, fz = zf - z0f;
    const int x0 = (int)x0f, y0 = (int)y0f, z0 = (int)z0f;
    const float* tb = rt + (size_t)i * 8 * 32768;
#pragma unroll
    for (int dzc = 0; dzc < 2; ++dzc)
#pragma unroll
    for (int dyc = 0; dyc < 2; ++dyc)
#pragma unroll
    for (int dxc = 0; dxc < 2; ++dxc) {
      const int xi = x0 + dxc, yi = y0 + dyc, zi = z0 + dzc;
      if (xi < 0 || xi >= 32 || yi < 0 || yi >= 32 || zi < 0 || zi >= 32) continue;
      const float w = (dxc ? fx : 1.0f - fx) * (dyc ? fy : 1.0f - fy) * (dzc ? fz : 1.0f - fz);
      const int idx = (zi * 32 + yi) * 32 + xi;
#pragma unroll
      for (int ch = 0; ch < 8; ++ch) acc[ch] += tb[(size_t)ch * 32768 + idx] * w;
    }
  }
#pragma unroll
  for (int ch = 0; ch < 8; ++ch) {
    const size_t si = (((size_t)(b * NH_ + ch)) * NQ_ + q) * (size_t)NK_ + kk;
    scores[si] += acc[ch];
  }
}

// ===================== host-side launchers =====================
static inline void launch_gemm(hipStream_t s,
    const float* A, int lda, long long sAb, long long sAh,
    const float* Bm, int ldb, int transB, long long sBb, long long sBh,
    const float* bias, float* out, int ldo, long long sOb, long long sOh,
    const float* resid, int res_mode, int nq,
    int M, int N, int K, float alpha, int relu, int nh, int batch)
{
  const int tiles = (M / 16) * (N / 16);
  dim3 grid((tiles + 3) / 4, batch);
  gemm_wmma<<<grid, 128, 0, s>>>(A, lda, sAb, sAh, Bm, ldb, transB, sBb, sBh,
                                 bias, out, ldo, sOb, sOh, resid, res_mode, nq,
                                 M, N, K, alpha, relu, nh);
}

static inline void launch_gemm_tiled(hipStream_t s,
    const float* A, int lda, long long sAb, long long sAh,
    const float* Bm, int ldb, long long sBb, long long sBh,
    const float* bias, float* out, int ldo, long long sOb, long long sOh,
    const float* resid, int res_mode, int nq,
    int M, int N, int K, float alpha, int relu, int nh, int batch)
{
  dim3 grid((M / 128) * (N / 64), batch);
  gemm_wmma_tiled<<<grid, 256, 0, s>>>(A, lda, sAb, sAh, Bm, ldb, sBb, sBh,
                                       bias, out, ldo, sOb, sOh, resid, res_mode, nq,
                                       M, N, K, alpha, relu, nh);
}

extern "C" void kernel_launch(void* const* d_in, const int* in_sizes, int n_in,
                              void* d_out, int out_size, void* d_ws, size_t ws_size,
                              hipStream_t stream) {
  (void)in_sizes; (void)n_in; (void)out_size; (void)ws_size;
  const float* tgt    = (const float*)d_in[0];
  const float* memory = (const float*)d_in[1];
  const float* refp   = (const float*)d_in[2];
  const float* refang = (const float*)d_in[3];
  const float* excyz  = (const float*)d_in[4];
  const float* qpos   = (const float*)d_in[5];
  const float* sa_wq = (const float*)d_in[6];  const float* sa_bq = (const float*)d_in[7];
  const float* sa_wk = (const float*)d_in[8];  const float* sa_bk = (const float*)d_in[9];
  const float* sa_wv = (const float*)d_in[10]; const float* sa_bv = (const float*)d_in[11];
  const float* sa_wp = (const float*)d_in[12]; const float* sa_bp = (const float*)d_in[13];
  const float* ca_wq = (const float*)d_in[14]; const float* ca_bq = (const float*)d_in[15];
  const float* ca_wk = (const float*)d_in[16]; const float* ca_bk = (const float*)d_in[17];
  const float* ca_wv = (const float*)d_in[18]; const float* ca_bv = (const float*)d_in[19];
  const float* ca_wp = (const float*)d_in[20]; const float* ca_bp = (const float*)d_in[21];
  const float* cpb_w1 = (const float*)d_in[22];
  const float* cpb_b1 = (const float*)d_in[23];
  const float* cpb_w2 = (const float*)d_in[24];
  const float* ln1_g = (const float*)d_in[25]; const float* ln1_b = (const float*)d_in[26];
  const float* ln2_g = (const float*)d_in[27]; const float* ln2_b = (const float*)d_in[28];
  const float* ln3_g = (const float*)d_in[29]; const float* ln3_b = (const float*)d_in[30];
  const float* ff_w1 = (const float*)d_in[31]; const float* ff_b1 = (const float*)d_in[32];
  const float* ff_w2 = (const float*)d_in[33]; const float* ff_b2 = (const float*)d_in[34];
  float* outp = (float*)d_out;

  float* w = (float*)d_ws;
  size_t o = 0;
  float* t2    = w + o; o += (size_t)B_ * NQ_ * C_;
  float* qk    = w + o; o += (size_t)B_ * NQ_ * C_;
  float* qbuf  = w + o; o += (size_t)B_ * NQ_ * C_;
  float* kbuf  = w + o; o += (size_t)B_ * NK_ * HD_;
  float* vbuf  = w + o; o += (size_t)B_ * NK_ * HD_;
  float* xbuf  = w + o; o += (size_t)B_ * NQ_ * C_;
  float* tgt1  = w + o; o += (size_t)B_ * NQ_ * C_;
  float* tgt2  = w + o; o += (size_t)B_ * NQ_ * C_;
  float* memT  = w + o; o += (size_t)B_ * NK_ * C_;
  float* rpet  = w + o; o += (size_t)8 * NH_ * 32768;
  float* cstab = w + o; o += (size_t)B_ * NQ_ * 2;
  float* hid   = w + o; o += (size_t)B_ * NQ_ * DFF_;
  float* scores= w + o; o += (size_t)B_ * NH_ * NQ_ * NK_;

  const float scale = 0.17677669529663687f;  // HD^-0.5
  const int MBQ = B_ * NQ_;
  const int MBK = B_ * NK_;

  transpose_mem<<<(MBK * C_) / 256, 256, 0, stream>>>(memory, memT);
  cos_sin_kernel<<<(MBQ + 255) / 256, 256, 0, stream>>>(refang, cstab);
  cpb_kernel<<<dim3(32768 / 256, 8), 256, 0, stream>>>(cpb_w1, cpb_b1, cpb_w2, rpet);

  // ================= phase 1: shared-KV self-attention =================
  layernorm_qk<<<MBQ, 256, 0, stream>>>(tgt, ln1_g, ln1_b, qpos, t2, qk);
  launch_gemm_tiled(stream, qk, C_, 0, 0, sa_wq, C_, 0, 0, sa_bq,
                    qbuf, C_, 0, 0, nullptr, 0, 0, MBQ, C_, C_, scale, 0, 1, 1);
  launch_gemm(stream, qk, C_, 0, 0, sa_wk, C_, 1, 0, 0, sa_bk,
              kbuf, HD_, 0, 0, nullptr, 0, 0, MBQ, HD_, C_, 1.0f, 0, 1, 1);
  launch_gemm(stream, t2, C_, 0, 0, sa_wv, C_, 1, 0, 0, sa_bv,
              vbuf, HD_, 0, 0, nullptr, 0, 0, MBQ, HD_, C_, 1.0f, 0, 1, 1);
  launch_gemm_tiled(stream, qbuf, C_, (long long)NQ_ * C_, HD_,
                    kbuf, HD_, (long long)NQ_ * HD_, 0, nullptr,
                    scores, NQ_, (long long)NH_ * NQ_ * NQ_, (long long)NQ_ * NQ_,
                    nullptr, 0, 0, NQ_, NQ_, HD_, 1.0f, 0, NH_, B_ * NH_);
  softmax_rows<<<B_ * NH_ * NQ_, 256, 0, stream>>>(scores, NQ_);
  launch_gemm(stream, scores, NQ_, (long long)NH_ * NQ_ * NQ_, (long long)NQ_ * NQ_,
              vbuf, HD_, 0, (long long)NQ_ * HD_, 0, nullptr,
              xbuf, C_, (long long)NQ_ * C_, HD_,
              nullptr, 0, 0, NQ_, HD_, NQ_, 1.0f, 0, NH_, B_ * NH_);
  launch_gemm_tiled(stream, xbuf, C_, 0, 0, sa_wp, C_, 0, 0, sa_bp,
                    tgt1, C_, 0, 0, tgt, 1, NQ_, MBQ, C_, C_, 1.0f, 0, 1, 1);

  // ================= phase 2: global cross-attention + RPE =================
  layernorm_qk<<<MBQ, 256, 0, stream>>>(tgt1, ln2_g, ln2_b, qpos, t2, qk);
  launch_gemm_tiled(stream, qk, C_, 0, 0, ca_wq, C_, 0, 0, ca_bq,
                    qbuf, C_, 0, 0, nullptr, 0, 0, MBQ, C_, C_, scale, 0, 1, 1);
  launch_gemm(stream, memT, C_, 0, 0, ca_wk, C_, 1, 0, 0, ca_bk,
              kbuf, HD_, 0, 0, nullptr, 0, 0, MBK, HD_, C_, 1.0f, 0, 1, 1);
  launch_gemm(stream, memT, C_, 0, 0, ca_wv, C_, 1, 0, 0, ca_bv,
              vbuf, HD_, 0, 0, nullptr, 0, 0, MBK, HD_, C_, 1.0f, 0, 1, 1);
  launch_gemm_tiled(stream, qbuf, C_, (long long)NQ_ * C_, HD_,
                    kbuf, HD_, (long long)NK_ * HD_, 0, nullptr,
                    scores, NK_, (long long)NH_ * NQ_ * NK_, (long long)NQ_ * NK_,
                    nullptr, 0, 0, NQ_, NK_, HD_, 1.0f, 0, NH_, B_ * NH_);
  rpe_add<<<(B_ * NQ_ * NK_) / 256, 256, 0, stream>>>(scores, refp, cstab, excyz, rpet);
  softmax_rows<<<B_ * NH_ * NQ_, 256, 0, stream>>>(scores, NK_);
  launch_gemm(stream, scores, NK_, (long long)NH_ * NQ_ * NK_, (long long)NQ_ * NK_,
              vbuf, HD_, 0, (long long)NK_ * HD_, 0, nullptr,
              xbuf, C_, (long long)NQ_ * C_, HD_,
              nullptr, 0, 0, NQ_, HD_, NK_, 1.0f, 0, NH_, B_ * NH_);
  launch_gemm_tiled(stream, xbuf, C_, 0, 0, ca_wp, C_, 0, 0, ca_bp,
                    tgt2, C_, 0, 0, tgt1, 1, NQ_, MBQ, C_, C_, 1.0f, 0, 1, 1);

  // ================= phase 3: FFN =================
  layernorm_qk<<<MBQ, 256, 0, stream>>>(tgt2, ln3_g, ln3_b, nullptr, t2, nullptr);
  launch_gemm_tiled(stream, t2, C_, 0, 0, ff_w1, C_, 0, 0, ff_b1,
                    hid, DFF_, 0, 0, nullptr, 0, 0, MBQ, DFF_, C_, 1.0f, 1, 1, 1);
  launch_gemm_tiled(stream, hid, DFF_, 0, 0, ff_w2, DFF_, 0, 0, ff_b2,
                    outp, C_, 0, 0, tgt2, 1, NQ_, MBQ, C_, DFF_, 1.0f, 0, 1, 1);
}
